// TFSwinSelfAttention_73942156968345
// MI455X (gfx1250) — matrix operand order repaired and checked by reading, not compiled, not run on hardware
//
#include <hip/hip_runtime.h>
#include <hip/hip_bf16.h>
#include <math.h>

typedef _Float16 f16;
typedef __attribute__((ext_vector_type(16))) _Float16 v16h;
typedef __attribute__((ext_vector_type(8)))  float    v8f;

#define WS 7
#define NTOK 49          // window tokens
#define NPAD 64          // padded tokens (M/K padding for 16x16x32 tiles)
#define NH 8
#define HD 32
#define DMODEL 256
#define NW 64
#define SROW 64          // S row stride (floats)

__device__ __forceinline__ v8f wmma_f16(v16h a, v16h b, v8f c) {
  return __builtin_amdgcn_wmma_f32_16x16x32_f16(
      /*neg_a=*/false, a, /*neg_b=*/false, b,
      /*c_mod=*/(short)0, c, /*reuse_a=*/false, /*reuse_b=*/false);
}

// A fragment: 16x32 (MxK), row-major source base[m*ld + k].
// Layout used here (contiguous k): lanes 0-15 -> M=lane, elements {0..15}=K kb..kb+15,
// where kb = k0 + (lane>=16 ? 16 : 0).
__device__ __forceinline__ v16h load_a(const f16* base, int ld, int m0, int k0, int lane) {
  const f16* p = base + (m0 + (lane & 15)) * ld + k0 + ((lane >> 4) << 4);
  v16h a;
#pragma unroll
  for (int i = 0; i < 16; ++i) a[i] = p[i];
  return a;
}

// B fragment: 32x16 (KxN), K-major source base[n*ld + k]  (i.e. columns of B contiguous).
// Layout used here (contiguous k, as load_a): lane = N column (lanes 0-15 K=kb..kb+15 with kb=k0, lanes 16-31 kb=k0+16).
__device__ __forceinline__ v16h load_b_kmajor(const f16* base, int ld, int k0, int n0, int lane) {
  const f16* p = base + (n0 + (lane & 15)) * ld + k0 + ((lane >> 4) << 4);
  v16h b;
#pragma unroll
  for (int i = 0; i < 16; ++i) b[i] = p[i];
  return b;
}

// B fragment: 32x16 (KxN), N-major source base[k*ld + n] (row-major B, strided gather).
__device__ __forceinline__ v16h load_b_nmajor(const f16* base, int ld, int k0, int n0, int lane) {
  int n  = n0 + (lane & 15);
  int kb = k0 + ((lane >> 4) << 4);
  v16h b;
#pragma unroll
  for (int i = 0; i < 16; ++i) b[i] = base[(kb + i) * ld + n];
  return b;
}

// ---------------- pre-kernel: transpose weights to f16 K-major ----------------
// out[mat][n][k] = (f16) w_mat[k][n];  3 matrices of 256x256.
__global__ __launch_bounds__(256) void swin_wt_prep(
    const float* __restrict__ wq, const float* __restrict__ wk,
    const float* __restrict__ wv, f16* __restrict__ wt) {
  int idx = blockIdx.x * 256 + threadIdx.x;          // 3*256*256 total
  int mat = idx >> 16;
  int n   = (idx >> 8) & 255;
  int k   = idx & 255;
  const float* w = (mat == 0) ? wq : (mat == 1) ? wk : wv;
  wt[idx] = (f16)w[k * DMODEL + n];
}

// ---------------- fused window-attention kernel: 1 block = 1 window ----------------
__global__ __launch_bounds__(256) void swin_attn(
    const float* __restrict__ x,        // [B,49,256]
    const float* __restrict__ amask,    // [64,49,49]
    const float* __restrict__ bq, const float* __restrict__ bk,
    const float* __restrict__ bvv,
    const float* __restrict__ relt,     // [169,8]
    const f16*   __restrict__ wt,       // [3][256 n][256 k] f16 K-major
    float* __restrict__ out) {          // [B,49,256]
  extern __shared__ char smem[];
  f16*   Xs = (f16*)smem;                        // 64x256
  f16*   Qs = Xs + NPAD * DMODEL;                // 64x256
  f16*   Ks = Qs + NPAD * DMODEL;                // 64x256
  f16*   Vs = Ks + NPAD * DMODEL;                // 64x256
  float* Ss = (float*)(Vs + NPAD * DMODEL);      // 8 x (49x64) f32
  f16*   Ps = (f16*)(Ss + NH * NTOK * SROW);     // 8 x (64x64) f16

  const int b    = blockIdx.x;
  const int tid  = threadIdx.x;
  const int lane = tid & 31;
  const int wave = tid >> 5;

  // ---- phase 0: stage X as f16 into LDS, zero pad rows 49..63 ----
  const float* xb = x + (size_t)b * NTOK * DMODEL;
  for (int i = tid; i < NPAD * DMODEL; i += 256) {
    int r = i >> 8;
    Xs[i] = (r < NTOK) ? (f16)xb[r * DMODEL + (i & 255)] : (f16)0.0f;
  }
  __syncthreads();

  // ---- phase 1: Q/K/V = X @ W + bias  (192 output tiles over 8 waves) ----
  for (int t = wave; t < 192; t += 8) {
    int mat  = t >> 6;                 // 0=Q 1=K 2=V
    int tile = t & 63;
    int m0 = (tile >> 4) << 4;
    int n0 = (tile & 15) << 4;
    const f16*   w    = wt + mat * DMODEL * DMODEL;
    const float* bias = (mat == 0) ? bq : (mat == 1) ? bk : bvv;
    f16*         dst  = (mat == 0) ? Qs : (mat == 1) ? Ks : Vs;
    float bcol = bias[n0 + (lane & 15)];
    v8f c;
#pragma unroll
    for (int r = 0; r < 8; ++r) c[r] = bcol;    // bias depends on column only
#pragma unroll
    for (int k0 = 0; k0 < DMODEL; k0 += 32) {
      v16h a = load_a(Xs, DMODEL, m0, k0, lane);
      v16h bf = load_b_kmajor(w, DMODEL, k0, n0, lane);
      c = wmma_f16(a, bf, c);
    }
    int n = n0 + (lane & 15);
    int mb = m0 + ((lane >> 4) << 3);
#pragma unroll
    for (int r = 0; r < 8; ++r) dst[(mb + r) * DMODEL + n] = (f16)c[r];
  }
  __syncthreads();

  // ---- from here each wave privately owns head h = wave ----
  const int h = wave;
  const f16* Qh = Qs + h * HD;
  const f16* Kh = Ks + h * HD;
  const f16* Vh = Vs + h * HD;
  float*     Sh = Ss + h * NTOK * SROW;
  f16*       Ph = Ps + h * NPAD * NPAD;
  const int  wi = b & (NW - 1);
  const float scale = 0.17677669529663687f;     // 1/sqrt(32)

  // ---- phase 2: S = scale * Q Kᵀ + rel_bias + mask  (16 tiles, K=32) ----
  for (int t = 0; t < 16; ++t) {
    int m0 = (t >> 2) << 4;
    int n0 = (t & 3) << 4;
    v16h a  = load_a(Qh, DMODEL, m0, 0, lane);
    v16h bf = load_b_kmajor(Kh, DMODEL, 0, n0, lane);   // B = Kᵀ: column n = K row n
    v8f c = {};
    c = wmma_f16(a, bf, c);
    int n  = n0 + (lane & 15);
    int mb = m0 + ((lane >> 4) << 3);
#pragma unroll
    for (int r = 0; r < 8; ++r) {
      int m = mb + r;
      if (m < NTOK) {
        float val;
        if (n < NTOK) {
          int qh_ = m / WS, qw_ = m % WS, kh_ = n / WS, kw_ = n % WS;
          int ridx = (qh_ - kh_ + WS - 1) * (2 * WS - 1) + (qw_ - kw_ + WS - 1);
          val = c[r] * scale + relt[ridx * NH + h]
              + amask[(wi * NTOK + m) * NTOK + n];
        } else {
          val = -3.0e38f;                       // pad keys: ignored by softmax
        }
        Sh[m * SROW + n] = val;
      }
    }
  }

  // ---- phase 3: row softmax -> P (f16), zero pad columns 49..63 ----
  for (int r = 0; r < NTOK; ++r) {
    float v0 = Sh[r * SROW + lane];
    int   c1 = lane + 32;
    float v1 = (c1 < NTOK) ? Sh[r * SROW + c1] : -3.0e38f;
    float mx = fmaxf(v0, v1);
#pragma unroll
    for (int off = 16; off; off >>= 1) mx = fmaxf(mx, __shfl_xor(mx, off, 32));
    float e0 = __expf(v0 - mx);
    float e1 = (c1 < NTOK) ? __expf(v1 - mx) : 0.0f;
    float s = e0 + e1;
#pragma unroll
    for (int off = 16; off; off >>= 1) s += __shfl_xor(s, off, 32);
    float inv = 1.0f / s;
    Ph[r * NPAD + lane] = (f16)(e0 * inv);
    Ph[r * NPAD + c1]   = (f16)((c1 < NTOK) ? e1 * inv : 0.0f);
  }

  // ---- phase 4: ctx = P @ V  (8 tiles, K=64 padded) ----
  for (int t = 0; t < 8; ++t) {
    int m0 = (t >> 1) << 4;
    int n0 = (t & 1) << 4;
    v8f c = {};
#pragma unroll
    for (int k0 = 0; k0 < NPAD; k0 += 32) {
      v16h a  = load_a(Ph, NPAD, m0, k0, lane);
      v16h bf = load_b_nmajor(Vh, DMODEL, k0, n0, lane);
      c = wmma_f16(a, bf, c);
    }
    int n  = n0 + (lane & 15);
    int mb = m0 + ((lane >> 4) << 3);
#pragma unroll
    for (int r = 0; r < 8; ++r) {
      int m = mb + r;
      if (m < NTOK)
        out[((size_t)b * NTOK + m) * DMODEL + h * HD + n] = c[r];
    }
  }
}

extern "C" void kernel_launch(void* const* d_in, const int* in_sizes, int n_in,
                              void* d_out, int out_size, void* d_ws, size_t ws_size,
                              hipStream_t stream) {
  const float* x     = (const float*)d_in[0];   // hidden_states [B,49,256]
  const float* amask = (const float*)d_in[1];   // attention_mask [64,49,49]
  const float* wq    = (const float*)d_in[2];
  const float* bq    = (const float*)d_in[3];
  const float* wk    = (const float*)d_in[4];
  const float* bk    = (const float*)d_in[5];
  const float* wv    = (const float*)d_in[6];
  const float* bv    = (const float*)d_in[7];
  const float* relt  = (const float*)d_in[8];   // [169,8]
  float* out = (float*)d_out;

  int B = in_sizes[0] / (NTOK * DMODEL);        // 4096
  f16* wt = (f16*)d_ws;                         // 3*256*256 f16 = 384 KB

  // weight transpose + f32->f16 (3*65536 elements, 768 blocks of 256)
  swin_wt_prep<<<dim3(3 * DMODEL * DMODEL / 256), dim3(256), 0, stream>>>(wq, wk, wv, wt);

  // LDS: X,Q,K,V (4 * 64*256 f16) + S (8 * 49*64 f32) + P (8 * 64*64 f16) = 296960 B
  size_t shmem = (size_t)4 * NPAD * DMODEL * sizeof(f16)
               + (size_t)NH * NTOK * SROW * sizeof(float)
               + (size_t)NH * NPAD * NPAD * sizeof(f16);

  swin_attn<<<dim3(B), dim3(256), shmem, stream>>>(
      x, amask, bq, bk, bv, relt, wt, out);
}
